// ModulatedSparseTransformerMMDiTBlock_14465449853514
// MI455X (gfx1250) — matrix operand order repaired and checked by reading, not compiled, not run on hardware
//
#include <hip/hip_runtime.h>
#include <hip/hip_bf16.h>
#include <math.h>
#include <stdint.h>

// ---------------------------------------------------------------------------
// CDNA5 (gfx1250) wave32 WMMA types
// ---------------------------------------------------------------------------
typedef __attribute__((ext_vector_type(16))) __bf16 v16bf;
typedef __attribute__((ext_vector_type(8)))  float  v8f;

#define WMMA_BF16(a, b, c) \
  __builtin_amdgcn_wmma_f32_16x16x32_bf16(false, (a), false, (b), (short)0, (c), false, false)

union FragU { v16bf v; uint4 q[2]; };

__device__ __forceinline__ v16bf ld_frag(const void* p0, const void* p1) {
  FragU f;
  f.q[0] = *(const uint4*)p0;
  f.q[1] = *(const uint4*)p1;
  return f.v;
}

__device__ __forceinline__ v8f zero8() {
  v8f z;
#pragma unroll
  for (int i = 0; i < 8; ++i) z[i] = 0.0f;
  return z;
}

// ---------------------------------------------------------------------------
// Elementwise helpers
// ---------------------------------------------------------------------------
__global__ void f2bf_kernel(const float* __restrict__ s, __bf16* __restrict__ d, int n) {
  for (int i = blockIdx.x * blockDim.x + threadIdx.x; i < n; i += gridDim.x * blockDim.x)
    d[i] = (__bf16)s[i];
}

__global__ void silu_kernel(const float* __restrict__ s, float* __restrict__ d, int n) {
  int i = blockIdx.x * blockDim.x + threadIdx.x;
  if (i < n) {
    float x = s[i];
    d[i] = x / (1.0f + expf(-x));
  }
}

// m = silu(mod) @ w_adaln + b_adaln   (M = 2, scalar is fine)
__global__ void adaln_kernel(const float* __restrict__ sm, const float* __restrict__ w,
                             const float* __restrict__ bias, float* __restrict__ m,
                             int C, int CO) {
  int j = blockIdx.x * blockDim.x + threadIdx.x;
  if (j >= 2 * CO) return;
  int b = j / CO, col = j % CO;
  float acc = bias[col];
  for (int k = 0; k < C; ++k)
    acc += sm[b * C + k] * w[(size_t)k * CO + col];
  m[j] = acc;
}

// out = base + val[row-mapped] * g   (used for attn residual and final outputs)
__global__ void resgate_kernel(const float* __restrict__ base,
                               const float* __restrict__ val, int Sval, int rowOff,
                               const float* __restrict__ g, int bStride,
                               float* __restrict__ out, int R, int C, int total) {
  for (int i = blockIdx.x * blockDim.x + threadIdx.x; i < total; i += gridDim.x * blockDim.x) {
    int c = i % C;
    int row = i / C;
    int b = row / R, r = row % R;
    out[i] = base[i] + val[(size_t)(b * Sval + rowOff + r) * C + c] * g[b * bStride + c];
  }
}

// ---------------------------------------------------------------------------
// LayerNorm + AdaLN modulate -> bf16 (C == 1024, one block per row)
// ---------------------------------------------------------------------------
__global__ __launch_bounds__(256) void ln_mod_kernel(
    const float* __restrict__ src, int Rsrc,
    const float* __restrict__ sc, const float* __restrict__ sh, int bStride,
    __bf16* __restrict__ dst, int Sdst, int rowOff, int C) {
  const int row = blockIdx.x;
  const int b = row / Rsrc, r = row % Rsrc;
  const float* x = src + (size_t)row * C;
  const int tid = threadIdx.x;
  float v[4], s = 0.0f, ss = 0.0f;
#pragma unroll
  for (int i = 0; i < 4; ++i) {
    v[i] = x[tid + i * 256];
    s += v[i];
    ss += v[i] * v[i];
  }
#pragma unroll
  for (int msk = 1; msk < 32; msk <<= 1) {
    s += __shfl_xor(s, msk);
    ss += __shfl_xor(ss, msk);
  }
  __shared__ float as_[8], ass_[8];
  if ((tid & 31) == 0) { as_[tid >> 5] = s; ass_[tid >> 5] = ss; }
  __syncthreads();
  float S1 = 0.0f, S2 = 0.0f;
#pragma unroll
  for (int i = 0; i < 8; ++i) { S1 += as_[i]; S2 += ass_[i]; }
  const float inv = 1.0f / (float)C;
  const float mu = S1 * inv;
  const float var = S2 * inv - mu * mu;
  const float rstd = rsqrtf(var + 1e-6f);
  __bf16* o = dst + (size_t)(b * Sdst + rowOff + r) * C;
#pragma unroll
  for (int i = 0; i < 4; ++i) {
    const int c = tid + i * 256;
    const float val = (v[i] - mu) * rstd * (1.0f + sc[b * bStride + c]) + sh[b * bStride + c];
    o[c] = (__bf16)val;
  }
}

// ---------------------------------------------------------------------------
// Generic bf16 WMMA GEMM: C = act(A[MxK] @ B[KxN] + bias)
// Block tile 128x128x32, 8 waves (4x2), each wave 32x64 (2x4 WMMA tiles).
// Software pipelined: next k-tile is loaded into registers while the current
// LDS tile feeds WMMAs; weight stream prefetched ahead into GL2.
// Running aPtr/bPtr pointers keep global addresses as immediate offsets.
// A staged row-major [128][40], B staged transposed [128][40] in LDS so that
// both fragment types load as two ds_load_b128 per lane (ISA 16-bit layouts).
// act: 0 = none, 1 = gelu(tanh), 2 = silu. outF / outH optional.
// ---------------------------------------------------------------------------
__global__ __launch_bounds__(256) void gemm_bf16_kernel(
    const __bf16* __restrict__ A, const __bf16* __restrict__ Bw,
    const float* __restrict__ bias,
    float* __restrict__ outF, __bf16* __restrict__ outH,
    int M, int N, int K, int act) {
  __shared__ __align__(16) __bf16 sA[128][40];
  __shared__ __align__(16) __bf16 sB[128][40];  // holds B^T tile: [n][k]
  const int tid = threadIdx.x;
  const int lane = tid & 31, w = tid >> 5;
  const int wm = w >> 1, wn = w & 1;
  const int half = lane >> 4, l16 = lane & 15;
  const int rb = blockIdx.y * 128, cb = blockIdx.x * 128;

  v8f acc[2][4];
#pragma unroll
  for (int t = 0; t < 2; ++t)
#pragma unroll
    for (int u = 0; u < 4; ++u) acc[t][u] = zero8();

  const int arow = tid >> 1, aseg = tid & 1;
  const int bn4 = (tid & 31) * 4, bk0 = tid >> 5;

  // running global pointers (strength-reduced addressing)
  const __bf16* aPtr = A + (size_t)(rb + arow) * K + aseg * 16;
  const __bf16* bPtr = Bw + (size_t)bk0 * N + cb + bn4;
  const size_t bStep = (size_t)8 * N;  // 8 k-rows between the 4 per-thread loads

  // register staging buffers (next k-tile)
  uint4 ra0, ra1;
  __bf16 rbv[16];

  auto loadTile = [&]() {
    const uint4* g = (const uint4*)aPtr;
    ra0 = g[0];
    ra1 = g[1];
#pragma unroll
    for (int it = 0; it < 4; ++it) {
      const __bf16* gb = bPtr + (size_t)it * bStep;
      rbv[it * 4 + 0] = gb[0];
      rbv[it * 4 + 1] = gb[1];
      rbv[it * 4 + 2] = gb[2];
      rbv[it * 4 + 3] = gb[3];
    }
    aPtr += 32;
    bPtr += 4 * bStep;  // advance one 32-k tile
  };

  loadTile();

  for (int k0 = 0; k0 < K; k0 += 32) {
    __syncthreads();
    // commit staged tile to LDS
    *(uint4*)&sA[arow][aseg * 16] = ra0;
    *(uint4*)&sA[arow][aseg * 16 + 8] = ra1;
#pragma unroll
    for (int it = 0; it < 4; ++it) {
      const int kk = bk0 + it * 8;
      sB[bn4 + 0][kk] = rbv[it * 4 + 0];
      sB[bn4 + 1][kk] = rbv[it * 4 + 1];
      sB[bn4 + 2][kk] = rbv[it * 4 + 2];
      sB[bn4 + 3][kk] = rbv[it * 4 + 3];
    }
    // start next tile's global loads (overlap with WMMA below)
    if (k0 + 32 < K) loadTile();
    // pull the weight stream ahead into GL2 (global_prefetch_b8); bPtr now
    // points at tile k0+64, so +8*bStep targets tile k0+128.
    if (k0 + 128 < K)
      __builtin_prefetch(bPtr + (size_t)8 * bStep, 0, 0);
    __syncthreads();

    v16bf aF[2], bF[4];
#pragma unroll
    for (int t = 0; t < 2; ++t) {
      const char* p = (const char*)&sA[wm * 32 + t * 16 + l16][0] + half * 16;
      aF[t] = ld_frag(p, p + 32);
    }
#pragma unroll
    for (int u = 0; u < 4; ++u) {
      const char* p = (const char*)&sB[wn * 64 + u * 16 + l16][0] + half * 16;
      bF[u] = ld_frag(p, p + 32);
    }
#pragma unroll
    for (int t = 0; t < 2; ++t)
#pragma unroll
      for (int u = 0; u < 4; ++u)
        acc[t][u] = WMMA_BF16(aF[t], bF[u], acc[t][u]);
  }

#pragma unroll
  for (int u = 0; u < 4; ++u) {
    const int col = cb + wn * 64 + u * 16 + l16;
    const float bv = bias ? bias[col] : 0.0f;
#pragma unroll
    for (int t = 0; t < 2; ++t) {
#pragma unroll
      for (int r = 0; r < 8; ++r) {
        float x = acc[t][u][r] + bv;
        if (act == 1) {
          x = 0.5f * x * (1.0f + tanhf(0.7978845608028654f * (x + 0.044715f * x * x * x)));
        } else if (act == 2) {
          x = x / (1.0f + __expf(-x));
        }
        const size_t idx = (size_t)(rb + wm * 32 + t * 16 + 8 * half + r) * N + col;
        if (outF) outF[idx] = x;
        if (outH) outH[idx] = (__bf16)x;
      }
    }
  }
}

// ---------------------------------------------------------------------------
// Flash attention over joint sequence. qkv: (B*S, 3C) bf16. out: (B*S, C) bf16.
// Block = (b, h, 128 query rows); 8 waves x 16 query rows each.
// Key chunk = 64; K tile [key][d], V^T tile [d][key], per-wave P buffer in LDS.
// ---------------------------------------------------------------------------
__global__ __launch_bounds__(256) void attn_kernel(
    const __bf16* __restrict__ qkv, __bf16* __restrict__ outH, int S, int C) {
  __shared__ __align__(16) __bf16 sK[64][72];
  __shared__ __align__(16) __bf16 sV[64][72];       // V^T : [d][key]
  __shared__ __align__(16) __bf16 sP[8][16][72];    // per-wave probabilities
  const int tid = threadIdx.x, lane = tid & 31, w = tid >> 5;
  const int half = lane >> 4, l16 = lane & 15;
  const int b = blockIdx.z, h = blockIdx.y;
  const int q0 = blockIdx.x * 128 + w * 16;
  const int C3 = 3 * C;

  v16bf qF[2];
  {
    const __bf16* qp = qkv + (size_t)(b * S + q0 + l16) * C3 + h * 64;
#pragma unroll
    for (int kc = 0; kc < 2; ++kc) {
      const char* p = (const char*)(qp + kc * 32) + half * 16;
      qF[kc] = ld_frag(p, p + 32);
    }
  }

  v8f o[4];
#pragma unroll
  for (int v = 0; v < 4; ++v) o[v] = zero8();
  float mr[8], lr[8];
#pragma unroll
  for (int r = 0; r < 8; ++r) { mr[r] = -3.0e38f; lr[r] = 0.0f; }

  // running pointers for K and V^T staging
  const __bf16* kPtr = qkv + (size_t)(b * S + (tid >> 2)) * C3 + C + h * 64 + (tid & 3) * 16;
  const __bf16* vPtr = qkv + (size_t)(b * S + (tid >> 4)) * C3 + 2 * C + h * 64 + (tid & 15) * 4;
  const size_t chunkStep = (size_t)64 * C3;
  const size_t vRowStep = (size_t)16 * C3;
  const int vd0 = (tid & 15) * 4, vk0 = tid >> 4;

  for (int j = 0; j < S; j += 64) {
    __syncthreads();
    {  // stage K tile (64 keys x 64 d)
      const uint4* g = (const uint4*)kPtr;
      uint4 v0 = g[0], v1 = g[1];
      *(uint4*)&sK[tid >> 2][(tid & 3) * 16] = v0;
      *(uint4*)&sK[tid >> 2][(tid & 3) * 16 + 8] = v1;
    }
#pragma unroll
    for (int it = 0; it < 4; ++it) {  // stage V^T tile (fixed 4 iterations, no guards)
      const int kk = vk0 + it * 16;
      const __bf16* g = vPtr + (size_t)it * vRowStep;
      __bf16 a0 = g[0], a1 = g[1], a2 = g[2], a3 = g[3];
      sV[vd0 + 0][kk] = a0;
      sV[vd0 + 1][kk] = a1;
      sV[vd0 + 2][kk] = a2;
      sV[vd0 + 3][kk] = a3;
    }
    kPtr += chunkStep;
    vPtr += chunkStep;
    __syncthreads();

    // scores: 16 q-rows x 64 keys
    v8f st[4];
#pragma unroll
    for (int u = 0; u < 4; ++u) st[u] = zero8();
#pragma unroll
    for (int u = 0; u < 4; ++u)
#pragma unroll
      for (int kc = 0; kc < 2; ++kc) {
        const char* p = (const char*)&sK[u * 16 + l16][0] + kc * 64 + half * 16;
        v16bf kF = ld_frag(p, p + 32);
        st[u] = WMMA_BF16(qF[kc], kF, st[u]);
      }

    // online softmax (row = r + 8*half; 16-lane half-wave reductions)
#pragma unroll
    for (int r = 0; r < 8; ++r) {
      float mx = fmaxf(fmaxf(st[0][r], st[1][r]), fmaxf(st[2][r], st[3][r])) * 0.125f;
      mx = fmaxf(mx, __shfl_xor(mx, 1));
      mx = fmaxf(mx, __shfl_xor(mx, 2));
      mx = fmaxf(mx, __shfl_xor(mx, 4));
      mx = fmaxf(mx, __shfl_xor(mx, 8));
      const float mn = fmaxf(mr[r], mx);
      const float alpha = __expf(mr[r] - mn);
      mr[r] = mn;
      float s = 0.0f;
#pragma unroll
      for (int u = 0; u < 4; ++u) {
        float p = __expf(st[u][r] * 0.125f - mn);
        st[u][r] = p;
        s += p;
      }
      s += __shfl_xor(s, 1);
      s += __shfl_xor(s, 2);
      s += __shfl_xor(s, 4);
      s += __shfl_xor(s, 8);
      lr[r] = lr[r] * alpha + s;
#pragma unroll
      for (int v = 0; v < 4; ++v) o[v][r] *= alpha;
    }

    // bounce P through wave-private LDS to get A-fragment layout
#pragma unroll
    for (int u = 0; u < 4; ++u)
#pragma unroll
      for (int r = 0; r < 8; ++r)
        sP[w][8 * half + r][u * 16 + l16] = (__bf16)st[u][r];
    asm volatile("s_wait_dscnt 0x0" ::: "memory");  // same-wave DS in-order; drain stores

    v16bf pF[2];
#pragma unroll
    for (int kc = 0; kc < 2; ++kc) {
      const char* p = (const char*)&sP[w][l16][0] + kc * 64 + half * 16;
      pF[kc] = ld_frag(p, p + 32);
    }
#pragma unroll
    for (int v = 0; v < 4; ++v)
#pragma unroll
      for (int kc = 0; kc < 2; ++kc) {
        const char* p = (const char*)&sV[v * 16 + l16][0] + kc * 64 + half * 16;
        v16bf vF = ld_frag(p, p + 32);
        o[v] = WMMA_BF16(pF[kc], vF, o[v]);
      }
  }

#pragma unroll
  for (int r = 0; r < 8; ++r) {
    const float inv = 1.0f / lr[r];
#pragma unroll
    for (int v = 0; v < 4; ++v)
      outH[(size_t)(b * S + q0 + 8 * half + r) * C + h * 64 + v * 16 + l16] =
          (__bf16)(o[v][r] * inv);
  }
}

// ---------------------------------------------------------------------------
// Host orchestration
// ---------------------------------------------------------------------------
extern "C" void kernel_launch(void* const* d_in, const int* in_sizes, int n_in,
                              void* d_out, int out_size, void* d_ws, size_t ws_size,
                              hipStream_t stream) {
  (void)in_sizes; (void)n_in; (void)out_size; (void)ws_size;
  const int Bb = 2, Nn = 2048, T = 256, C = 1024, H = 4096, NH = 16;
  const int S = Nn + T;  // 2304

  const float* x       = (const float*)d_in[0];
  const float* mod     = (const float*)d_in[1];
  const float* cond    = (const float*)d_in[2];
  const float* w_adaln = (const float*)d_in[3];
  const float* b_adaln = (const float*)d_in[4];
  const float* w_qkv   = (const float*)d_in[5];
  const float* b_qkv   = (const float*)d_in[6];
  const float* w_proj  = (const float*)d_in[7];
  const float* b_proj  = (const float*)d_in[8];
  const float* w_s1    = (const float*)d_in[9];
  const float* b_s1    = (const float*)d_in[10];
  const float* w_s2    = (const float*)d_in[11];
  const float* b_s2    = (const float*)d_in[12];
  const float* w_d1    = (const float*)d_in[13];
  const float* b_d1    = (const float*)d_in[14];
  const float* w_d2    = (const float*)d_in[15];
  const float* b_d2    = (const float*)d_in[16];

  char* ws = (char*)d_ws;
  size_t off = 0;
  auto alloc = [&](size_t bytes) -> void* {
    void* p = ws + off;
    off = (off + bytes + 255) & ~(size_t)255;
    return p;
  };

  float*  sm      = (float*)alloc((size_t)Bb * C * 4);
  float*  m       = (float*)alloc((size_t)Bb * 6 * C * 4);
  __bf16* wq_bf   = (__bf16*)alloc((size_t)C * 3 * C * 2);
  __bf16* wp_bf   = (__bf16*)alloc((size_t)C * C * 2);
  __bf16* ws1_bf  = (__bf16*)alloc((size_t)C * H * 2);
  __bf16* ws2_bf  = (__bf16*)alloc((size_t)H * C * 2);
  __bf16* wd1_bf  = (__bf16*)alloc((size_t)C * H * 2);
  __bf16* wd2_bf  = (__bf16*)alloc((size_t)H * C * 2);
  __bf16* joint   = (__bf16*)alloc((size_t)Bb * S * C * 2);
  __bf16* qkv     = (__bf16*)alloc((size_t)Bb * S * 3 * C * 2);
  __bf16* attn_o  = (__bf16*)alloc((size_t)Bb * S * C * 2);
  float*  proj_f  = (float*)alloc((size_t)Bb * S * C * 4);
  float*  x1      = (float*)alloc((size_t)Bb * Nn * C * 4);
  float*  c1      = (float*)alloc((size_t)Bb * T * C * 4);
  __bf16* x1ln    = (__bf16*)alloc((size_t)Bb * Nn * C * 2);
  __bf16* c1ln    = (__bf16*)alloc((size_t)Bb * T * C * 2);
  __bf16* hx      = (__bf16*)alloc((size_t)Bb * Nn * H * 2);
  __bf16* hc      = (__bf16*)alloc((size_t)Bb * T * H * 2);
  float*  mlpx    = (float*)alloc((size_t)Bb * Nn * C * 4);
  float*  mlpc    = (float*)alloc((size_t)Bb * T * C * 4);

  const float* sh_msa = m;
  const float* sc_msa = m + C;
  const float* g_msa  = m + 2 * C;
  const float* sh_mlp = m + 3 * C;
  const float* sc_mlp = m + 4 * C;
  const float* g_mlp  = m + 5 * C;
  const int bStride = 6 * C;

  // 1) modulation vector
  silu_kernel<<<(Bb * C + 255) / 256, 256, 0, stream>>>(mod, sm, Bb * C);
  adaln_kernel<<<(Bb * 6 * C + 255) / 256, 256, 0, stream>>>(sm, w_adaln, b_adaln, m, C, 6 * C);

  // 2) weights -> bf16
  f2bf_kernel<<<2048, 256, 0, stream>>>(w_qkv, wq_bf, C * 3 * C);
  f2bf_kernel<<<2048, 256, 0, stream>>>(w_proj, wp_bf, C * C);
  f2bf_kernel<<<2048, 256, 0, stream>>>(w_s1, ws1_bf, C * H);
  f2bf_kernel<<<2048, 256, 0, stream>>>(w_s2, ws2_bf, H * C);
  f2bf_kernel<<<2048, 256, 0, stream>>>(w_d1, wd1_bf, C * H);
  f2bf_kernel<<<2048, 256, 0, stream>>>(w_d2, wd2_bf, H * C);

  // 3) LN + modulate -> joint (bf16)
  ln_mod_kernel<<<Bb * Nn, 256, 0, stream>>>(x, Nn, sc_msa, sh_msa, bStride, joint, S, 0, C);
  ln_mod_kernel<<<Bb * T, 256, 0, stream>>>(cond, T, sc_msa, sh_msa, bStride, joint, S, Nn, C);

  // 4) qkv = joint @ w_qkv + b_qkv  (bf16 out)
  gemm_bf16_kernel<<<dim3(3 * C / 128, Bb * S / 128), 256, 0, stream>>>(
      joint, wq_bf, b_qkv, nullptr, qkv, Bb * S, 3 * C, C, 0);

  // 5) joint attention
  attn_kernel<<<dim3(S / 128, NH, Bb), 256, 0, stream>>>(qkv, attn_o, S, C);

  // 6) proj (f32 out)
  gemm_bf16_kernel<<<dim3(C / 128, Bb * S / 128), 256, 0, stream>>>(
      attn_o, wp_bf, b_proj, proj_f, nullptr, Bb * S, C, C, 0);

  // 7) attention residual with g_msa
  resgate_kernel<<<(Bb * Nn * C + 255) / 256, 256, 0, stream>>>(
      x, proj_f, S, 0, g_msa, bStride, x1, Nn, C, Bb * Nn * C);
  resgate_kernel<<<(Bb * T * C + 255) / 256, 256, 0, stream>>>(
      cond, proj_f, S, Nn, g_msa, bStride, c1, T, C, Bb * T * C);

  // 8) LN + modulate for MLPs
  ln_mod_kernel<<<Bb * Nn, 256, 0, stream>>>(x1, Nn, sc_mlp, sh_mlp, bStride, x1ln, Nn, 0, C);
  ln_mod_kernel<<<Bb * T, 256, 0, stream>>>(c1, T, sc_mlp, sh_mlp, bStride, c1ln, T, 0, C);

  // 9) x-MLP: gelu(x1ln @ w_s1 + b_s1) @ w_s2 + b_s2
  gemm_bf16_kernel<<<dim3(H / 128, Bb * Nn / 128), 256, 0, stream>>>(
      x1ln, ws1_bf, b_s1, nullptr, hx, Bb * Nn, H, C, 1);
  gemm_bf16_kernel<<<dim3(C / 128, Bb * Nn / 128), 256, 0, stream>>>(
      hx, ws2_bf, b_s2, mlpx, nullptr, Bb * Nn, C, H, 0);

  // 10) cond-MLP: silu(c1ln @ w_d1 + b_d1) @ w_d2 + b_d2
  gemm_bf16_kernel<<<dim3(H / 128, Bb * T / 128), 256, 0, stream>>>(
      c1ln, wd1_bf, b_d1, nullptr, hc, Bb * T, H, C, 2);
  gemm_bf16_kernel<<<dim3(C / 128, Bb * T / 128), 256, 0, stream>>>(
      hc, wd2_bf, b_d2, mlpc, nullptr, Bb * T, C, H, 0);

  // 11) final residuals with g_mlp -> d_out (x then cond, concatenated)
  float* out_x = (float*)d_out;
  float* out_c = (float*)d_out + (size_t)Bb * Nn * C;
  resgate_kernel<<<(Bb * Nn * C + 255) / 256, 256, 0, stream>>>(
      x1, mlpx, Nn, 0, g_mlp, bStride, out_x, Nn, C, Bb * Nn * C);
  resgate_kernel<<<(Bb * T * C + 255) / 256, 256, 0, stream>>>(
      c1, mlpc, T, 0, g_mlp, bStride, out_c, T, C, Bb * T * C);
}
